// DecoderOnlyBlock_23201413333572
// MI455X (gfx1250) — compile-verified, matrix operand order checked by reading, not run on hardware
//
#include <hip/hip_runtime.h>

// ---------------- types ----------------
typedef __bf16 bf16_t;
typedef bf16_t v16bf __attribute__((ext_vector_type(16)));
typedef float  v8f   __attribute__((ext_vector_type(8)));

union FragBF { unsigned u[8]; v16bf v; };

static __device__ inline v8f vzero8() {
    v8f z = {0.f,0.f,0.f,0.f,0.f,0.f,0.f,0.f};
    return z;
}

static __device__ inline unsigned short f2bf(float x) {
    unsigned u = __float_as_uint(x);
    unsigned r = (u + 0x7FFFu + ((u >> 16) & 1u)) >> 16;
    return (unsigned short)r;
}

// Gather one 16x32 bf16 A/B fragment (row-major source, 16 uints per 32-elem row
// chunk). rowbase points at this lane's row, at the current k-step, in uint units.
// Layout per CDNA5 ISA: lanes 0-15 / 16-31 split K halves.
static __device__ inline void load_frag(FragBF& f, const unsigned* rowbase, int half) {
#pragma unroll
    for (int v = 0; v < 4; ++v) {
        f.u[v]     = rowbase[half * 4 + v];
        f.u[4 + v] = rowbase[8 + half * 4 + v];
    }
}

static __device__ inline v8f wmma_bf16(const FragBF& a, const FragBF& b, v8f c) {
    return __builtin_amdgcn_wmma_f32_16x16x32_bf16(false, a.v, false, b.v,
                                                   (short)0, c, false, false);
}

// CDNA5 async global->LDS copy, 16 bytes per lane (ASYNCcnt-tracked).
static __device__ inline void async_copy_b128(unsigned lds_byte_off, const void* gaddr) {
    asm volatile("global_load_async_to_lds_b128 %0, %1, off"
                 :: "v"(lds_byte_off), "v"(gaddr) : "memory");
}
static __device__ inline void wait_async0() {
    asm volatile("s_wait_asynccnt 0x0" ::: "memory");
}

// ---------------- constants ----------------
#define BB   4
#define SS   4096
#define DD   768
#define HH   8
#define HD   96
#define DFF  3072
#define NBLK 64
#define BSZ  64
#define NKB  5
#define MTOT (BB * SS)   // 16384

// ---------------- fp32 -> bf16, transposed: src[K][N] -> dst[N][K] ----------------
__global__ __launch_bounds__(256) void cvt_bf16_t_kernel(const float* __restrict__ src,
                                                         unsigned short* __restrict__ dst,
                                                         int K, int N) {
    int idx = blockIdx.x * 256 + threadIdx.x;
    if (idx < K * N) {
        int k = idx / N, n = idx % N;
        dst[(size_t)n * K + k] = f2bf(src[idx]);
    }
}

// ---------------- LayerNorm (fp32 in -> bf16 out) ----------------
__global__ __launch_bounds__(256) void ln_bf16_kernel(const float* __restrict__ x,
                                                      const float* __restrict__ g,
                                                      const float* __restrict__ bta,
                                                      unsigned short* __restrict__ out) {
    __shared__ float red[8];
    const int row = blockIdx.x;
    const float* xr = x + (size_t)row * DD;
    float v0[3];
    float s = 0.f;
#pragma unroll
    for (int j = 0; j < 3; ++j) { v0[j] = xr[threadIdx.x + 256 * j]; s += v0[j]; }
#pragma unroll
    for (int o = 16; o > 0; o >>= 1) s += __shfl_xor(s, o, 32);
    if ((threadIdx.x & 31) == 0) red[threadIdx.x >> 5] = s;
    __syncthreads();
    float tot = 0.f;
#pragma unroll
    for (int k = 0; k < 8; ++k) tot += red[k];
    const float mean = tot * (1.f / (float)DD);
    __syncthreads();
    float vs = 0.f;
#pragma unroll
    for (int j = 0; j < 3; ++j) { float d = v0[j] - mean; vs += d * d; }
#pragma unroll
    for (int o = 16; o > 0; o >>= 1) vs += __shfl_xor(vs, o, 32);
    if ((threadIdx.x & 31) == 0) red[threadIdx.x >> 5] = vs;
    __syncthreads();
    float vt = 0.f;
#pragma unroll
    for (int k = 0; k < 8; ++k) vt += red[k];
    const float inv = rsqrtf(vt * (1.f / (float)DD) + 1e-5f);
#pragma unroll
    for (int j = 0; j < 3; ++j) {
        int c = threadIdx.x + 256 * j;
        float y = (v0[j] - mean) * inv * g[c] + bta[c];
        out[(size_t)row * DD + c] = f2bf(y);
    }
}

// ---------------- WMMA GEMM: out[M,N] = A[M,K](bf16) @ Wt[N,K](bf16)^T + bias ----
// BM=128 (8 waves x 16 rows), BN=128 (8 col tiles/wave), BK=32.
// Double-buffered LDS staged via global_load_async_to_lds_b128.
template <bool OB16, bool RELU, bool RES>
__global__ __launch_bounds__(256) void gemm_bf16_kernel(const unsigned short* __restrict__ A,
                                                        const unsigned short* __restrict__ Wt,
                                                        const float* __restrict__ bias,
                                                        const float* __restrict__ res,
                                                        void* __restrict__ out,
                                                        int N, int K) {
    // per buffer: A 128x32 bf16 (2048 uints) + Wt 128x32 bf16 (2048 uints)
    __shared__ unsigned ldsU[2 * 4096];

    const int tid  = threadIdx.x;
    const int w    = tid >> 5;
    const int L    = tid & 31;
    const int half = L >> 4;
    const int l16  = L & 15;
    const int mBase = blockIdx.x * 128;
    const int nBase = blockIdx.y * 128;

    const unsigned ldsBase = (unsigned)(size_t)(&ldsU[0]);

    v8f acc[8];
#pragma unroll
    for (int t = 0; t < 8; ++t) acc[t] = vzero8();

    // stage one BK slice (A tile + W tile) into buffer `buf` with async copies:
    // 512 chunks of 16B per tile, 2 chunks per thread per tile.
    auto stage = [&](int buf, int k0) {
        const unsigned base = ldsBase + buf * 16384u;
#pragma unroll
        for (int j = 0; j < 2; ++j) {
            int idx  = j * 256 + tid;
            int row  = idx >> 2;
            int quad = idx & 3;
            const char* ga = (const char*)A  + ((size_t)(mBase + row) * K + k0) * 2 + quad * 16;
            const char* gw = (const char*)Wt + ((size_t)(nBase + row) * K + k0) * 2 + quad * 16;
            async_copy_b128(base + idx * 16, ga);
            async_copy_b128(base + 8192u + idx * 16, gw);
        }
    };

    const int nk = K >> 5;
    stage(0, 0);
    for (int it = 0; it < nk; ++it) {
        wait_async0();        // own async writes for current buffer done
        __syncthreads();      // all waves' writes done; also WAR fence for prev reads
        if (it + 1 < nk) stage((it + 1) & 1, (it + 1) << 5);

        const unsigned* bufA = ldsU + (it & 1) * 4096;
        const unsigned* bufW = bufA + 2048;
        FragBF a;
        load_frag(a, bufA + (w * 16 + l16) * 16, half);
#pragma unroll
        for (int t = 0; t < 8; ++t) {
            FragBF b;
            load_frag(b, bufW + (t * 16 + l16) * 16, half);
            acc[t] = wmma_bf16(a, b, acc[t]);
        }
    }

    // epilogue: bias (+relu) (+residual), bf16 or fp32 store
#pragma unroll
    for (int t = 0; t < 8; ++t) {
        int n = nBase + t * 16 + l16;
        float bv = bias[n];
#pragma unroll
        for (int r = 0; r < 8; ++r) {
            int m = mBase + w * 16 + half * 8 + r;
            float vv = acc[t][r] + bv;
            if (RELU) vv = vv > 0.f ? vv : 0.f;
            size_t off = (size_t)m * N + n;
            if (RES) vv += res[off];
            if (OB16) ((unsigned short*)out)[off] = f2bf(vv);
            else      ((float*)out)[off] = vv;
        }
    }
}

// ---------------- fused BigBird attention per (b, h, block) ----------------
// Dyn LDS: scores fp32 [64][320] | probs bf16 [64][320] | V^T bf16 [96][320]
#define S_BYTES  (64 * 320 * 4)
#define P_BYTES  (64 * 320 * 2)
#define VT_BYTES (96 * 320 * 2)
#define ATTN_LDS (S_BYTES + P_BYTES + VT_BYTES)

__global__ __launch_bounds__(256) void bigbird_attn_kernel(const unsigned short* __restrict__ qb,
                                                           const unsigned short* __restrict__ kb,
                                                           const unsigned short* __restrict__ vb,
                                                           const int* __restrict__ kb_idx,
                                                           const unsigned char* __restrict__ amask,
                                                           unsigned short* __restrict__ ctxb) {
    extern __shared__ char smem[];
    float*          Ssc = (float*)smem;
    unsigned short* P   = (unsigned short*)(smem + S_BYTES);
    unsigned short* VT  = (unsigned short*)(smem + S_BYTES + P_BYTES);

    const int tid  = threadIdx.x;
    const int w    = tid >> 5;
    const int L    = tid & 31;
    const int half = L >> 4;
    const int l16  = L & 15;

    const int blk = blockIdx.x;
    const int i = blk & 63;
    const int h = (blk >> 6) & 7;
    const int b = blk >> 9;

    int kbi[NKB];
#pragma unroll
    for (int s = 0; s < NKB; ++s) kbi[s] = kb_idx[i * NKB + s];

    // Phase 0: stage gathered V^T (d-major rows) into LDS
    for (int idx = tid; idx < HD * (NKB * BSZ); idx += 256) {
        int d = idx % HD, p = idx / HD;
        int seq = kbi[p >> 6] * BSZ + (p & 63);
        VT[d * 320 + p] = vb[((size_t)(b * SS + seq)) * DD + h * HD + d];
    }

    // Phase 1: scores = Q K^T (WMMA), scale + mask -> LDS fp32
    const unsigned* qg = (const unsigned*)qb;
    const unsigned* kg = (const unsigned*)kb;
    const int rt   = w >> 1;       // row tile 0..3
    const int csel = w & 1;        // half of the 20 col tiles
    const int mrow = rt * 16 + l16;
    const unsigned* qrow = qg + ((size_t)(b * SS + i * BSZ + mrow) * DD + h * HD) / 2;

    FragBF qf[3];
#pragma unroll
    for (int ks = 0; ks < 3; ++ks) load_frag(qf[ks], qrow + ks * 16, half);

    const float scale = 0.1020620726159658f;  // 1/sqrt(96)
    for (int tt = 0; tt < 10; ++tt) {
        int t = csel * 10 + tt;
        int n = t * 16 + l16;                 // 0..319
        int s = n >> 6;
        int kseq = kbi[s] * BSZ + (n & 63);
        const unsigned* krow = kg + ((size_t)(b * SS + kseq) * DD + h * HD) / 2;
        v8f acc = vzero8();
#pragma unroll
        for (int ks = 0; ks < 3; ++ks) {
            FragBF bf;
            load_frag(bf, krow + ks * 16, half);
            acc = wmma_bf16(qf[ks], bf, acc);
        }
        int p = n & 63;
#pragma unroll
        for (int r = 0; r < 8; ++r) {
            int m = rt * 16 + half * 8 + r;
            unsigned char mk = amask[(((size_t)i * BSZ + m) * NKB + s) * BSZ + p];
            Ssc[m * 320 + n] = mk ? acc[r] * scale : -1e9f;
        }
    }
    __syncthreads();

    // Phase 2: row softmax -> bf16 probs
    if (tid < 64) {
        const int row = tid;
        float mx = -3.0e38f;
        for (int n = 0; n < 320; ++n) mx = fmaxf(mx, Ssc[row * 320 + n]);
        float sum = 0.f;
        for (int n = 0; n < 320; ++n) sum += __expf(Ssc[row * 320 + n] - mx);
        float inv = 1.f / sum;
        for (int n = 0; n < 320; ++n)
            P[row * 320 + n] = f2bf(__expf(Ssc[row * 320 + n] - mx) * inv);
    }
    __syncthreads();

    // Phase 3: ctx = P @ V (WMMA), 4x6 tiles over 8 waves
    const unsigned* Pu = (const unsigned*)P;
    const unsigned* Vu = (const unsigned*)VT;
#pragma unroll
    for (int j = 0; j < 3; ++j) {
        int id = w * 3 + j;        // 0..23
        int rt2 = id / 6, ct = id % 6;
        int m_l = rt2 * 16 + l16;
        int d_l = ct * 16 + l16;
        v8f acc = vzero8();
        for (int ks = 0; ks < 10; ++ks) {
            FragBF af, bf;
            load_frag(af, Pu + m_l * 160 + ks * 16, half);
            load_frag(bf, Vu + d_l * 160 + ks * 16, half);
            acc = wmma_bf16(af, bf, acc);
        }
#pragma unroll
        for (int r = 0; r < 8; ++r) {
            int m = i * BSZ + rt2 * 16 + half * 8 + r;
            int d = h * HD + ct * 16 + l16;
            ctxb[((size_t)(b * SS + m)) * DD + d] = f2bf(acc[r]);
        }
    }
}

// ---------------- host ----------------
extern "C" void kernel_launch(void* const* d_in, const int* in_sizes, int n_in,
                              void* d_out, int out_size, void* d_ws, size_t ws_size,
                              hipStream_t stream) {
    const float* x    = (const float*)d_in[0];
    const float* Wq   = (const float*)d_in[1];
    const float* bq   = (const float*)d_in[2];
    const float* Wk   = (const float*)d_in[3];
    const float* bk   = (const float*)d_in[4];
    const float* Wv   = (const float*)d_in[5];
    const float* bv   = (const float*)d_in[6];
    const float* Wo   = (const float*)d_in[7];
    const float* bo   = (const float*)d_in[8];
    const float* ln_g = (const float*)d_in[9];
    const float* ln_b = (const float*)d_in[10];
    const float* W1   = (const float*)d_in[11];
    const float* b1   = (const float*)d_in[12];
    const float* W2   = (const float*)d_in[13];
    const float* b2   = (const float*)d_in[14];
    const int*   kbi  = (const int*)d_in[15];
    const unsigned char* amask = (const unsigned char*)d_in[16];
    float* out = (float*)d_out;

    char* ws = (char*)d_ws;
    size_t off = 0;
    auto alloc = [&](size_t bytes) -> void* {
        void* p = ws + off;
        off = (off + bytes + 255) & ~(size_t)255;
        return p;
    };
    unsigned short* hb   = (unsigned short*)alloc((size_t)MTOT * DD * 2);
    unsigned short* qbuf = (unsigned short*)alloc((size_t)MTOT * DD * 2);
    unsigned short* kbuf = (unsigned short*)alloc((size_t)MTOT * DD * 2);
    unsigned short* vbuf = (unsigned short*)alloc((size_t)MTOT * DD * 2);
    unsigned short* ctxb = (unsigned short*)alloc((size_t)MTOT * DD * 2);
    float*          x1   = (float*)alloc((size_t)MTOT * DD * 4);
    unsigned short* h2b  = (unsigned short*)alloc((size_t)MTOT * DD * 2);
    unsigned short* f1b  = (unsigned short*)alloc((size_t)MTOT * DFF * 2);
    unsigned short* Wqt  = (unsigned short*)alloc((size_t)DD * DD * 2);
    unsigned short* Wkt  = (unsigned short*)alloc((size_t)DD * DD * 2);
    unsigned short* Wvt  = (unsigned short*)alloc((size_t)DD * DD * 2);
    unsigned short* Wot  = (unsigned short*)alloc((size_t)DD * DD * 2);
    unsigned short* W1t  = (unsigned short*)alloc((size_t)DD * DFF * 2);
    unsigned short* W2t  = (unsigned short*)alloc((size_t)DFF * DD * 2);

    const int nW  = DD * DD;
    const int nW1 = DD * DFF;
    cvt_bf16_t_kernel<<<(nW + 255) / 256, 256, 0, stream>>>(Wq, Wqt, DD, DD);
    cvt_bf16_t_kernel<<<(nW + 255) / 256, 256, 0, stream>>>(Wk, Wkt, DD, DD);
    cvt_bf16_t_kernel<<<(nW + 255) / 256, 256, 0, stream>>>(Wv, Wvt, DD, DD);
    cvt_bf16_t_kernel<<<(nW + 255) / 256, 256, 0, stream>>>(Wo, Wot, DD, DD);
    cvt_bf16_t_kernel<<<(nW1 + 255) / 256, 256, 0, stream>>>(W1, W1t, DD, DFF);
    cvt_bf16_t_kernel<<<(nW1 + 255) / 256, 256, 0, stream>>>(W2, W2t, DFF, DD);

    // h = LN(x) -> bf16
    ln_bf16_kernel<<<MTOT, 256, 0, stream>>>(x, ln_g, ln_b, hb);

    // QKV projections
    dim3 g768(MTOT / 128, DD / 128);
    gemm_bf16_kernel<true, false, false><<<g768, 256, 0, stream>>>(hb, Wqt, bq, nullptr, qbuf, DD, DD);
    gemm_bf16_kernel<true, false, false><<<g768, 256, 0, stream>>>(hb, Wkt, bk, nullptr, kbuf, DD, DD);
    gemm_bf16_kernel<true, false, false><<<g768, 256, 0, stream>>>(hb, Wvt, bv, nullptr, vbuf, DD, DD);

    // BigBird attention
    bigbird_attn_kernel<<<BB * HH * NBLK, 256, ATTN_LDS, stream>>>(qbuf, kbuf, vbuf, kbi, amask, ctxb);

    // x1 = x + ctx @ Wo + bo
    gemm_bf16_kernel<false, false, true><<<g768, 256, 0, stream>>>(ctxb, Wot, bo, x, x1, DD, DD);

    // h2 = LN(x1) -> bf16
    ln_bf16_kernel<<<MTOT, 256, 0, stream>>>(x1, ln_g, ln_b, h2b);

    // f1 = relu(h2 @ W1 + b1) -> bf16
    dim3 gffn(MTOT / 128, DFF / 128);
    gemm_bf16_kernel<true, true, false><<<gffn, 256, 0, stream>>>(h2b, W1t, b1, nullptr, f1b, DFF, DD);

    // out = x1 + f1 @ W2 + b2
    gemm_bf16_kernel<false, false, true><<<g768, 256, 0, stream>>>(f1b, W2t, b2, x1, out, DD, DFF);

    (void)in_sizes; (void)n_in; (void)out_size; (void)ws_size;
}
